// lightgcl_frame_81432579932607
// MI455X (gfx1250) — matrix operand order repaired and checked by reading, not compiled, not run on hardware
//
#include <hip/hip_runtime.h>
#include <cstdint>
#include <cstddef>

#define NU 100000
#define NI 50000
#define DIM 64
#define QD 5
#define NNZ_E 1000000
#define BB 1024
#define TEMP 0.2f
#define LAM1 0.2f
#define LAM2 1e-5f
#define KEEPP 0.75f
#define LOG2E 1.4426950408889634f

typedef __attribute__((ext_vector_type(16))) _Float16 v16h;
typedef __attribute__((ext_vector_type(8)))  float    v8f;

union V16H { uint4 q[2]; v16h v; };

__device__ __forceinline__ float fast_exp2(float x) {
#if __has_builtin(__builtin_amdgcn_exp2f)
    return __builtin_amdgcn_exp2f(x);   // raw v_exp_f32, no libm range fixup
#else
    return exp2f(x);
#endif
}

__device__ __forceinline__ uint32_t hash32(uint32_t x) {
    x ^= x >> 16; x *= 0x7feb352dU;
    x ^= x >> 15; x *= 0x846ca68bU;
    x ^= x >> 16; return x;
}

__device__ __forceinline__ float block_reduce_sum256(float v) {
    __shared__ float sh[256];
    int t = threadIdx.x;
    sh[t] = v; __syncthreads();
    for (int s = 128; s > 0; s >>= 1) {
        if (t < s) sh[t] += sh[t + s];
        __syncthreads();
    }
    return sh[0];
}

// dst1 = dst2 = dst3 = src
__global__ void k_copy3(const float* __restrict__ src, float* a, float* b, float* c, int n) {
    int i = blockIdx.x * 256 + threadIdx.x;
    if (i < n) { float v = src[i]; a[i] = v; b[i] = v; c[i] = v; }
}

// dst += src
__global__ void k_add(float* __restrict__ dst, const float* __restrict__ src, int n) {
    int i = blockIdx.x * 256 + threadIdx.x;
    if (i < n) dst[i] += src[i];
}

// Y[out[e]] += drop(vals[e]) * X[gth[e]]   (one thread per (edge, 4 dims))
__global__ void k_spmm(const float* __restrict__ vals, const int* __restrict__ outIdx,
                       const int* __restrict__ gthIdx, const float* __restrict__ X,
                       float* __restrict__ Y, uint32_t seed) {
    int tid = blockIdx.x * 256 + threadIdx.x;
    if (tid >= NNZ_E * 16) return;
    int e = tid >> 4, dq = (tid & 15) << 2;
    uint32_t h = hash32((uint32_t)e * 2654435761u + seed * 0x9E3779B9u + 0x85EBCA6Bu);
    if ((h & 0xFFFFFFu) < (uint32_t)(KEEPP * 16777216.0f)) {
        float w = vals[e] * (1.0f / KEEPP);
        const float4 x = *(const float4*)(X + (size_t)gthIdx[e] * DIM + dq);
        float* y = Y + (size_t)outIdx[e] * DIM + dq;
        atomicAdd(y + 0, w * x.x);
        atomicAdd(y + 1, w * x.y);
        atomicAdd(y + 2, w * x.z);
        atomicAdd(y + 3, w * x.w);
    }
}

// W[q][d] = sum_n T[q*nrows+n] * X[n*64+d];  one block per (q,d)
__global__ void k_small_gemm(const float* __restrict__ T, const float* __restrict__ X,
                             float* __restrict__ W, int nrows) {
    int q = blockIdx.x >> 6, d = blockIdx.x & 63;
    float s = 0.f;
    for (int n = threadIdx.x; n < nrows; n += 256)
        s += T[(size_t)q * nrows + n] * X[(size_t)n * DIM + d];
    s = block_reduce_sum256(s);
    if (threadIdx.x == 0) W[blockIdx.x] = s;
}

// sG[n][d] += sum_q M[n*Q+q] * W[q*64+d]
__global__ void k_gacc(const float* __restrict__ M, const float* __restrict__ W,
                       float* __restrict__ sG, int nrows) {
    int i = blockIdx.x * 256 + threadIdx.x;
    if (i >= nrows * DIM) return;
    int n = i >> 6, d = i & 63;
    float acc = 0.f;
#pragma unroll
    for (int q = 0; q < QD; q++) acc += M[n * QD + q] * W[q * DIM + d];
    sG[i] += acc;
}

__global__ void k_tohalf(const float* __restrict__ x, _Float16* __restrict__ h, int n) {
    int i = blockIdx.x * 256 + threadIdx.x;
    if (i < n) h[i] = (_Float16)x[i];
}

// out[r*64+d] = (f16) S[idx(r)*64+d], idx(r) = r<n0 ? i0[r] : i1[r-n0]
__global__ void k_gatherh(const float* __restrict__ S, const int* __restrict__ i0,
                          const int* __restrict__ i1, int n0, int R, _Float16* __restrict__ out) {
    int i = blockIdx.x * 256 + threadIdx.x;
    if (i >= R * DIM) return;
    int r = i >> 6, d = i & 63;
    int id = (r < n0) ? i0[r] : i1[r - n0];
    out[i] = (_Float16)S[(size_t)id * DIM + d];
}

// ---- fused (A @ B^T)*scale -> exp2 -> row-sum via WMMA ----
// A: R x 64 f16 row-major; Bm: N x 64 f16 row-major (logical B columns).
// 8 waves per block; each wave owns one 16-row tile x disjoint 4-tile column chunks.
__device__ __forceinline__ void score_tile(const V16H& a0, const V16H& a1,
                                           const V16H& b0, const V16H& b1,
                                           float scale, float s[8]) {
    v8f c = {0.f, 0.f, 0.f, 0.f, 0.f, 0.f, 0.f, 0.f};
    c = __builtin_amdgcn_wmma_f32_16x16x32_f16(false, a0.v, false, b0.v, (short)0, c, false, false);
    c = __builtin_amdgcn_wmma_f32_16x16x32_f16(false, a1.v, false, b1.v, (short)0, c, false, false);
#pragma unroll
    for (int r = 0; r < 8; r++) s[r] += fast_exp2(c[r] * scale);
}

__global__ __launch_bounds__(256)
void k_score(const _Float16* __restrict__ A, const _Float16* __restrict__ Bm,
             float* __restrict__ rowsum, int ncols, float scale) {
    const int lane = threadIdx.x & 31;
    const int wid  = threadIdx.x >> 5;
    const int m = lane & 15, kh = lane >> 4;
    const int rt = blockIdx.y;

    // A tile per ISA 16-bit A(16x32) layout: lane holds row rt*16+m,
    // VGPR0-3 -> K = kh*8 + 0..7, VGPR4-7 -> K = 16 + kh*8 + 0..7
    const _Float16* arow = A + (size_t)(rt * 16 + m) * DIM;
    V16H a0, a1;
    a0.q[0] = *(const uint4*)(arow + kh * 8);
    a0.q[1] = *(const uint4*)(arow + 16 + kh * 8);
    a1.q[0] = *(const uint4*)(arow + 32 + kh * 8);
    a1.q[1] = *(const uint4*)(arow + 48 + kh * 8);

    float s[8];
#pragma unroll
    for (int r = 0; r < 8; r++) s[r] = 0.f;

    const int ntiles = ncols >> 4;
    const int stride = (int)gridDim.x * 8 * 4;     // all waves, 4 tiles each
    int ct = (blockIdx.x * 8 + wid) * 4;

    // main loop: 4 column tiles per iteration (16 b128 loads -> 8 WMMAs)
    for (; ct + 3 < ntiles; ct += stride) {
        // B tile per ISA 16-bit B(32x16) layout: lane = column, K = kh*8 + 0..7
        // and 16 + kh*8 + 0..7 (b0), plus 32 for b1
        const _Float16* base = Bm + ((size_t)ct * 16 + m) * DIM + kh * 8;
        V16H b0[4], b1[4];
#pragma unroll
        for (int t = 0; t < 4; t++) {
            const _Float16* brow = base + (size_t)t * 16 * DIM;
            b0[t].q[0] = *(const uint4*)(brow);
            b0[t].q[1] = *(const uint4*)(brow + 16);
            b1[t].q[0] = *(const uint4*)(brow + 32);
            b1[t].q[1] = *(const uint4*)(brow + 48);
        }
        if (ct + stride + 3 < ntiles)
            __builtin_prefetch(base + (size_t)stride * 16 * DIM, 0, 1);
#pragma unroll
        for (int t = 0; t < 4; t++)
            score_tile(a0, a1, b0[t], b1[t], scale, s);
    }
    // tail (uniform per wave; at most 3 tiles across the whole grid)
    for (; ct < ntiles; ct++) {
        const _Float16* brow = Bm + ((size_t)ct * 16 + m) * DIM + kh * 8;
        V16H b0, b1;
        b0.q[0] = *(const uint4*)(brow);
        b0.q[1] = *(const uint4*)(brow + 16);
        b1.q[0] = *(const uint4*)(brow + 32);
        b1.q[1] = *(const uint4*)(brow + 48);
        score_tile(a0, a1, b0, b1, scale, s);
    }

    // C layout: lanes 0-15 hold rows 0-7, lanes 16-31 hold rows 8-15; columns
    // stripe across the 16 lanes of each half -> reduce within halves.
#pragma unroll
    for (int r = 0; r < 8; r++)
        for (int off = 8; off >= 1; off >>= 1)
            s[r] += __shfl_xor(s[r], off, 32);
    if (m == 0) {
#pragma unroll
        for (int r = 0; r < 8; r++)
            atomicAdd(&rowsum[rt * 16 + kh * 8 + r], s[r]);
    }
}

__global__ void k_logsum(const float* __restrict__ x, int n, float* acc) {
    float v = 0.f;
    for (int i = blockIdx.x * 256 + threadIdx.x; i < n; i += gridDim.x * 256)
        v += logf(x[i] + 1e-8f);
    v = block_reduce_sum256(v);
    if (threadIdx.x == 0) atomicAdd(acc, v);
}

__global__ void k_pos(const float* __restrict__ G, const float* __restrict__ E,
                      const int* __restrict__ i0, const int* __restrict__ i1,
                      int n0, int R, float* acc) {
    int r = blockIdx.x * 256 + threadIdx.x;
    if (r >= R) return;
    int id = (r < n0) ? i0[r] : i1[r - n0];
    const float* g = G + (size_t)id * DIM;
    const float* e = E + (size_t)id * DIM;
    float dot = 0.f;
#pragma unroll
    for (int d = 0; d < DIM; d++) dot += g[d] * e[d];
    float v = fminf(5.0f, fmaxf(-5.0f, dot * (1.0f / TEMP)));
    atomicAdd(acc, v);
}

__global__ void k_bpr(const float* __restrict__ sEu, const float* __restrict__ sEi,
                      const int* __restrict__ uids, const int* __restrict__ pos,
                      const int* __restrict__ neg, float* acc) {
    int r = blockIdx.x * 256 + threadIdx.x;
    if (r >= BB) return;
    const float* u = sEu + (size_t)uids[r] * DIM;
    const float* p = sEi + (size_t)pos[r] * DIM;
    const float* n = sEi + (size_t)neg[r] * DIM;
    float dp = 0.f, dn = 0.f;
#pragma unroll
    for (int d = 0; d < DIM; d++) { dp += u[d] * p[d]; dn += u[d] * n[d]; }
    float diff = dp - dn;
    float ls = (diff >= 0.f) ? -log1pf(expf(-diff)) : (diff - log1pf(expf(diff)));
    atomicAdd(acc, ls);
}

__global__ void k_reg(const float* __restrict__ x, int n, float* acc) {
    float v = 0.f;
    for (int i = blockIdx.x * 256 + threadIdx.x; i < n; i += gridDim.x * 256)
        v += x[i] * x[i];
    v = block_reduce_sum256(v);
    if (threadIdx.x == 0) atomicAdd(acc, v);
}

__global__ void k_final(const float* __restrict__ scal, float* __restrict__ out) {
    float negu = scal[0] / (float)BB;
    float negi = scal[1] / (float)(2 * BB);
    float posu = scal[2] / (float)BB;
    float posi = scal[3] / (float)(2 * BB);
    float loss_s = -(posu + posi) + (negu + negi);
    float loss_r = -scal[4] / (float)BB;
    float loss_reg = LAM2 * scal[5];
    out[0] = loss_r + LAM1 * loss_s + loss_reg;
    out[1] = loss_r;
    out[2] = LAM1 * loss_s;
}

#define GS(n) (((n) + 255) / 256)

extern "C" void kernel_launch(void* const* d_in, const int* in_sizes, int n_in,
                              void* d_out, int out_size, void* d_ws, size_t ws_size,
                              hipStream_t stream) {
    (void)in_sizes; (void)n_in; (void)out_size; (void)ws_size;
    const float* E_u_0   = (const float*)d_in[0];
    const float* E_i_0   = (const float*)d_in[1];
    const float* u_mul_s = (const float*)d_in[2];
    const float* v_mul_s = (const float*)d_in[3];
    const float* ut      = (const float*)d_in[4];
    const float* vt      = (const float*)d_in[5];
    const float* adj_vals = (const float*)d_in[6];
    const int* adj_rows = (const int*)d_in[7];
    const int* adj_cols = (const int*)d_in[8];
    const int* uids = (const int*)d_in[9];
    const int* pos  = (const int*)d_in[10];
    const int* neg  = (const int*)d_in[11];
    float* out = (float*)d_out;

    float* ws = (float*)d_ws;
    size_t off = 0;
    auto alloc = [&](size_t nfloats) -> float* {
        float* p = ws + off;
        off += (nfloats + 63) & ~(size_t)63;
        return p;
    };
    float* Eu[2] = { alloc((size_t)NU * DIM), alloc((size_t)NU * DIM) };
    float* Ei[2] = { alloc((size_t)NI * DIM), alloc((size_t)NI * DIM) };
    float* sEu = alloc((size_t)NU * DIM);
    float* sEi = alloc((size_t)NI * DIM);
    float* sGu = alloc((size_t)NU * DIM);
    float* sGi = alloc((size_t)NI * DIM);
    float* W   = alloc(QD * DIM);
    float* rsu = alloc(BB);
    float* rsi = alloc(2 * BB);
    float* scal = alloc(64);
    _Float16* sEu_h = (_Float16*)alloc((size_t)NU * DIM / 2);
    _Float16* sEi_h = (_Float16*)alloc((size_t)NI * DIM / 2);
    _Float16* Au_h  = (_Float16*)alloc((size_t)BB * DIM / 2);
    _Float16* Ai_h  = (_Float16*)alloc((size_t)2 * BB * DIM / 2);

    // ---- init: E_cur = sE = sG = E_0 ----
    k_copy3<<<GS(NU * DIM), 256, 0, stream>>>(E_u_0, Eu[0], sEu, sGu, NU * DIM);
    k_copy3<<<GS(NI * DIM), 256, 0, stream>>>(E_i_0, Ei[0], sEi, sGi, NI * DIM);

    // ---- 2 propagation layers ----
    int cur = 0, nxt = 1;
    for (int l = 0; l < 2; l++) {
        hipMemsetAsync(Eu[nxt], 0, (size_t)NU * DIM * sizeof(float), stream);
        hipMemsetAsync(Ei[nxt], 0, (size_t)NI * DIM * sizeof(float), stream);
        // Z_u = drop(A) @ E_i ; Z_i = drop(A)^T @ E_u (independent dropout seeds)
        k_spmm<<<GS(NNZ_E * 16), 256, 0, stream>>>(adj_vals, adj_rows, adj_cols,
                                                   Ei[cur], Eu[nxt], (uint32_t)(2 * l));
        k_spmm<<<GS(NNZ_E * 16), 256, 0, stream>>>(adj_vals, adj_cols, adj_rows,
                                                   Eu[cur], Ei[nxt], (uint32_t)(2 * l + 1));
        // SVD branch: sG_u += u_mul_s @ (vt @ E_i); sG_i += v_mul_s @ (ut @ E_u)
        k_small_gemm<<<QD * DIM, 256, 0, stream>>>(vt, Ei[cur], W, NI);
        k_gacc<<<GS(NU * DIM), 256, 0, stream>>>(u_mul_s, W, sGu, NU);
        k_small_gemm<<<QD * DIM, 256, 0, stream>>>(ut, Eu[cur], W, NU);
        k_gacc<<<GS(NI * DIM), 256, 0, stream>>>(v_mul_s, W, sGi, NI);
        // running sums
        k_add<<<GS(NU * DIM), 256, 0, stream>>>(sEu, Eu[nxt], NU * DIM);
        k_add<<<GS(NI * DIM), 256, 0, stream>>>(sEi, Ei[nxt], NI * DIM);
        cur ^= 1; nxt ^= 1;
    }

    // ---- scoring ----
    hipMemsetAsync(rsu, 0, BB * sizeof(float), stream);
    hipMemsetAsync(rsi, 0, 2 * BB * sizeof(float), stream);
    hipMemsetAsync(scal, 0, 64 * sizeof(float), stream);

    k_tohalf<<<GS(NU * DIM), 256, 0, stream>>>(sEu, sEu_h, NU * DIM);
    k_tohalf<<<GS(NI * DIM), 256, 0, stream>>>(sEi, sEi_h, NI * DIM);
    k_gatherh<<<GS(BB * DIM), 256, 0, stream>>>(sGu, uids, uids, BB, BB, Au_h);
    k_gatherh<<<GS(2 * BB * DIM), 256, 0, stream>>>(sGi, pos, neg, BB, 2 * BB, Ai_h);

    // InfoNCE negative scores: fused WMMA GEMM + exp2 + row-sum
    const float scale = LOG2E / TEMP;
    k_score<<<dim3(30, BB / 16), 256, 0, stream>>>(Au_h, sEu_h, rsu, NU, scale);
    k_score<<<dim3(30, 2 * BB / 16), 256, 0, stream>>>(Ai_h, sEi_h, rsi, NI, scale);

    k_logsum<<<4, 256, 0, stream>>>(rsu, BB, scal + 0);
    k_logsum<<<8, 256, 0, stream>>>(rsi, 2 * BB, scal + 1);
    k_pos<<<GS(BB), 256, 0, stream>>>(sGu, sEu, uids, uids, BB, BB, scal + 2);
    k_pos<<<GS(2 * BB), 256, 0, stream>>>(sGi, sEi, pos, neg, BB, 2 * BB, scal + 3);
    k_bpr<<<GS(BB), 256, 0, stream>>>(sEu, sEi, uids, pos, neg, scal + 4);
    k_reg<<<1024, 256, 0, stream>>>(E_u_0, NU * DIM, scal + 5);
    k_reg<<<1024, 256, 0, stream>>>(E_i_0, NI * DIM, scal + 5);
    k_final<<<1, 1, 0, stream>>>(scal, out);
}
